// SSA_dwc_2628519985652
// MI455X (gfx1250) — matrix operand order repaired and checked by reading, not compiled, not run on hardware
//
#include <hip/hip_runtime.h>
#include <hip/hip_bf16.h>
#include <math.h>

// ---------------- problem constants ----------------
#define TT   4
#define BB   32
#define CC   512
#define HWN  256          // H*W
#define NHD  8            // heads
#define DHD  64           // head dim
#define TBM  (BB*CC*HWN)  // per-timestep element count = 4,194,304
#define CCN  (CC*HWN)

typedef __attribute__((ext_vector_type(16))) _Float16 v16h;
typedef __attribute__((ext_vector_type(8)))  float    v8f;

union V16H { uint4 q[2]; v16h v; };
union V8H  { uint4 u; _Float16 h[8]; };

// ---------------- WMMA helpers (CDNA5 wave32 layouts) ----------------
__device__ __forceinline__ v8f wmma_f16(v16h a, v16h b, v8f c) {
  return __builtin_amdgcn_wmma_f32_16x16x32_f16(false, a, false, b, (short)0, c,
                                                false, false);
}

// A fragment 16(M)x32(K), source row-major (M rows, K contiguous, ld halves).
// lanes 0-15: M=lane, halves0-7=K0..7, halves8-15=K16..23; lanes16-31: K+8.
__device__ __forceinline__ v16h frag_a(const _Float16* p, int ld, int lid) {
  int m  = lid & 15;
  int ko = (lid & 16) ? 8 : 0;
  const _Float16* r = p + (size_t)m * ld + ko;
  V16H u;
  u.q[0] = *(const uint4*)(r);
  u.q[1] = *(const uint4*)(r + 16);
  return u.v;
}

// B fragment 32(K)x16(N), source N-major: element (k,n) at p[n*ld + k].
// lane n=lid&15 holds K = kg..kg+7 and kg+16..kg+23 (kg=0/8) => two 16B loads.
__device__ __forceinline__ v16h frag_b_rows(const _Float16* p, int ld, int lid) {
  int n  = lid & 15;
  int kg = (lid & 16) ? 8 : 0;
  const _Float16* r = p + (size_t)n * ld + kg;
  V16H u;
  u.q[0] = *(const uint4*)(r);
  u.q[1] = *(const uint4*)(r + 16);
  return u.v;
}

// ---------------- prep kernels ----------------
__global__ __launch_bounds__(256) void k_prep_bn(
    const float* __restrict__ gamma, const float* __restrict__ beta,
    const float* __restrict__ mean, const float* __restrict__ var,
    const float* __restrict__ proj_b, const float* __restrict__ dwc_b,
    float* __restrict__ scale, float* __restrict__ bias) {
  int i = blockIdx.x * 256 + threadIdx.x;
  if (i < 5 * CC) {
    int set = i / CC, c = i % CC;
    float sc = gamma[i] * rsqrtf(var[i] + 1e-5f);
    float bi = beta[i] - mean[i] * sc;
    if (set == 3) bi += sc * proj_b[c];
    if (set == 4) bi += sc * dwc_b[c];
    scale[i] = sc;
    bias[i]  = bi;
  }
}

__global__ __launch_bounds__(256) void k_w2h(
    const float* __restrict__ wq, const float* __restrict__ wk,
    const float* __restrict__ wv, const float* __restrict__ wp,
    _Float16* __restrict__ w16) {
  int i = blockIdx.x * 256 + threadIdx.x;
  if (i < 4 * CC * CC) {
    int m = i >> 18, r = i & (CC * CC - 1);
    const float* src = (m == 0) ? wq : (m == 1) ? wk : (m == 2) ? wv : wp;
    w16[i] = (_Float16)src[r];
  }
}

// ---------------- input LIF + transpose to [t,b,n,c] spikes --------------
// grid (N/32, C/32, B); 256 thr; 32x32 tile transpose through LDS.
__global__ __launch_bounds__(256) void k_lif_in(const float* __restrict__ x,
                                                _Float16* __restrict__ xs) {
  __shared__ _Float16 tile[32][33];
  const int n0 = blockIdx.x * 32, c0 = blockIdx.y * 32, b = blockIdx.z;
  const int col = threadIdx.x & 31;
  const int r8  = threadIdx.x >> 5;  // 0..7
  float v[4] = {0.f, 0.f, 0.f, 0.f};
  for (int t = 0; t < TT; ++t) {
    const size_t tb = (size_t)t * TBM + (size_t)b * CCN;
#pragma unroll
    for (int s = 0; s < 4; ++s) {  // rows c_local = r8 + 8s, cols n (coalesced)
      float xt = x[tb + (size_t)(c0 + r8 + 8 * s) * HWN + n0 + col];
      float& vv = v[s];
      vv = 0.5f * (vv + xt);
      float sp = (vv >= 1.0f) ? 1.0f : 0.0f;
      vv *= (1.0f - sp);
      tile[r8 + 8 * s][col] = (_Float16)sp;  // tile[c_l][n_l]
    }
    __syncthreads();
#pragma unroll
    for (int s = 0; s < 4; ++s)  // rows n_local, cols c (coalesced)
      xs[tb + (size_t)(n0 + r8 + 8 * s) * CC + c0 + col] = tile[col][r8 + 8 * s];
    __syncthreads();
  }
}

// ---------------- fused GEMM + BN + LIF (spikes [n][c], no LDS) ----------
// out[t,b,n,d] = LIF_t( scale[d]*sum_c in[t,b,n,c]*w[d,c] + bias[d] )
// M=n, N=d. block 128 thr (4 waves 2x2); block tile 64x64; wave 32x32.
__global__ __launch_bounds__(128) void k_gemm_bn_lif(
    const _Float16* __restrict__ w16,   // [512,512] row-major (d,c)
    const _Float16* __restrict__ xin,   // [T,B,N,C] f16 spikes
    _Float16* __restrict__ sout,        // [T,B,N,C] f16 spikes
    const float* __restrict__ scale, const float* __restrict__ bias,
    float vth) {
  const int tid = threadIdx.x;
  const int lid = tid & 31;
  const int wid = tid >> 5;
  const int wm = wid >> 1, wn = wid & 1;
  const int bmn = blockIdx.x;  // n tiles: 0..3
  const int bnd = blockIdx.y;  // d tiles: 0..7
  const int b   = blockIdx.z;
  const int hi8 = (lid & 16) ? 8 : 0;
  const int nlo = lid & 15;

  const int nrow0 = bmn * 64 + wm * 32;  // wave's n rows
  const int dcol0 = bnd * 64 + wn * 32;  // wave's d cols
  // BN params are lane-indexed by d and t-invariant: hoist
  float sc[2], bi[2];
#pragma unroll
  for (int tj = 0; tj < 2; ++tj) {
    sc[tj] = scale[dcol0 + tj * 16 + nlo];
    bi[tj] = bias[dcol0 + tj * 16 + nlo];
  }

  float vstate[2][2][8];
#pragma unroll
  for (int i = 0; i < 2; ++i)
#pragma unroll
    for (int j = 0; j < 2; ++j)
#pragma unroll
      for (int r = 0; r < 8; ++r) vstate[i][j][r] = 0.f;

  for (int t = 0; t < TT; ++t) {
    v8f acc[2][2] = {};
    const _Float16* xb = xin + (size_t)(t * BB + b) * CCN;
    for (int kk = 0; kk < CC; kk += 32) {
      if (kk + 32 < CC)  // prefetch next spike chunk for this wave's rows
        __builtin_prefetch(xb + (size_t)(nrow0 + lid) * CC + kk + 32, 0, 1);
      v16h af[2], bf[2];
#pragma unroll
      for (int ti = 0; ti < 2; ++ti)  // A = spikes rows n (row-major over c)
        af[ti] = frag_a(xb + (size_t)(nrow0 + ti * 16) * CC + kk, CC, lid);
#pragma unroll
      for (int tj = 0; tj < 2; ++tj)  // B[k=c][d] = w[d,c]: N-major rows
        bf[tj] = frag_b_rows(w16 + (size_t)(dcol0 + tj * 16) * CC + kk, CC, lid);
#pragma unroll
      for (int ti = 0; ti < 2; ++ti)
#pragma unroll
        for (int tj = 0; tj < 2; ++tj)
          acc[ti][tj] = wmma_f16(af[ti], bf[tj], acc[ti][tj]);
    }
    // epilogue: BN + LIF, stores lane-contiguous over d
    _Float16* ob = sout + (size_t)(t * BB + b) * CCN;
#pragma unroll
    for (int ti = 0; ti < 2; ++ti)
#pragma unroll
      for (int tj = 0; tj < 2; ++tj)
#pragma unroll
        for (int r = 0; r < 8; ++r) {
          int ng = nrow0 + ti * 16 + r + hi8;
          int dg = dcol0 + tj * 16 + nlo;
          float y = sc[tj] * acc[ti][tj][r] + bi[tj];
          float& vs = vstate[ti][tj][r];
          vs = 0.5f * (vs + y);
          float s = (vs >= vth) ? 1.0f : 0.0f;
          ob[(size_t)ng * CC + dg] = (_Float16)s;
          vs *= (1.0f - s);
        }
  }
}

// ---------------- fused attention (spikes [n][c]) ----------------
// per (b,h): kvT[e][dd] = sum_n v[n,e]*k[n,dd]  (exact counts <=256)
//            o[n][e]    = 0.125*sum_dd q[n,dd]*kvT[e][dd] ; LIF(0.5)
__global__ __launch_bounds__(256) void k_attn(
    const _Float16* __restrict__ q_s, const _Float16* __restrict__ k_s,
    const _Float16* __restrict__ v_s, _Float16* __restrict__ o_s) {
  __shared__ __align__(16) _Float16 kT[64 * 40];   // k chunk [dd][32 n]
  __shared__ __align__(16) _Float16 vT[64 * 40];   // v chunk [e ][32 n]
  __shared__ __align__(16) _Float16 kvl[64 * 72];  // kvT [e][dd]
  const int tid = threadIdx.x;
  const int lid = tid & 31;
  const int wid = tid >> 5;  // 0..7
  const int b = blockIdx.x;
  const int h = blockIdx.y;
  const int hi8 = (lid & 16) ? 8 : 0;
  const int nlo = lid & 15;
  const int h64 = h * DHD;

  float vstate[2][4][8];  // (n-subtile, e-tile, reg)
#pragma unroll
  for (int i = 0; i < 2; ++i)
#pragma unroll
    for (int j = 0; j < 4; ++j)
#pragma unroll
      for (int r = 0; r < 8; ++r) vstate[i][j][r] = 0.f;

  for (int t = 0; t < TT; ++t) {
    const size_t tb = (size_t)(t * BB + b) * CCN;
    // ---- stage 1: kvT, K=256 over n; transpose-stage k/v chunks ----
    v8f acc1[4] = {};
    for (int kk = 0; kk < HWN; kk += 32) {
      {
        int nrow = tid >> 3;       // 0..31
        int c8 = (tid & 7) * 8;    // 8 halves along c
        V8H a, c;
        a.u = *(const uint4*)(k_s + tb + (size_t)(kk + nrow) * CC + h64 + c8);
        c.u = *(const uint4*)(v_s + tb + (size_t)(kk + nrow) * CC + h64 + c8);
#pragma unroll
        for (int q = 0; q < 8; ++q) {
          kT[(c8 + q) * 40 + nrow] = a.h[q];
          vT[(c8 + q) * 40 + nrow] = c.h[q];
        }
      }
      __syncthreads();
      if (wid < 4) {
        // A = v chunk rows e in [16*wid,+16), row-major over n
        v16h af = frag_a(vT + 16 * wid * 40, 40, lid);
#pragma unroll
        for (int tj = 0; tj < 4; ++tj) {
          // B[k=n][dd]: k chunk [dd][n] is N-major rows
          v16h bf = frag_b_rows(kT + (tj * 16) * 40, 40, lid);
          acc1[tj] = wmma_f16(af, bf, acc1[tj]);
        }
      }
      __syncthreads();
    }
    if (wid < 4) {  // store kvT[e][dd]; lane-contiguous over dd
#pragma unroll
      for (int tj = 0; tj < 4; ++tj)
#pragma unroll
        for (int r = 0; r < 8; ++r)
          kvl[(16 * wid + r + hi8) * 72 + tj * 16 + nlo] =
              (_Float16)acc1[tj][r];
    }
    __syncthreads();
    // ---- stage 2: o[n][e], K=64 over dd; wave -> n strip [32*wid,+32) ----
    v8f acc2[2][4] = {};
#pragma unroll
    for (int ks = 0; ks < 2; ++ks) {
      v16h af2[2];
#pragma unroll
      for (int ti = 0; ti < 2; ++ti)  // A = q rows n, row-major over dd
        af2[ti] = frag_a(q_s + tb + (size_t)(32 * wid + ti * 16) * CC + h64 +
                             ks * 32,
                         CC, lid);
#pragma unroll
      for (int tj = 0; tj < 4; ++tj) {
        // B[k=dd][e]: kvT [e][dd] is N-major rows
        v16h bf2 = frag_b_rows(kvl + (tj * 16) * 72 + ks * 32, 72, lid);
#pragma unroll
        for (int ti = 0; ti < 2; ++ti)
          acc2[ti][tj] = wmma_f16(af2[ti], bf2, acc2[ti][tj]);
      }
    }
    // epilogue: scale + LIF(0.5); stores lane-contiguous over e
#pragma unroll
    for (int ti = 0; ti < 2; ++ti)
#pragma unroll
      for (int tj = 0; tj < 4; ++tj)
#pragma unroll
        for (int r = 0; r < 8; ++r) {
          int ng = 32 * wid + ti * 16 + r + hi8;
          int eg = tj * 16 + nlo;
          float y = 0.125f * acc2[ti][tj][r];
          float& vs = vstate[ti][tj][r];
          vs = 0.5f * (vs + y);
          float s = (vs >= 0.5f) ? 1.0f : 0.0f;
          o_s[tb + (size_t)ng * CC + h64 + eg] = (_Float16)s;
          vs *= (1.0f - s);
        }
    __syncthreads();
  }
}

// ---------------- transpose spikes [t,b,n,c] -> [t,b,c,n] ----------------
__global__ __launch_bounds__(256) void k_tr_nc_cn(const _Float16* __restrict__ zin,
                                                  _Float16* __restrict__ zout) {
  __shared__ _Float16 tile[32][33];
  const int c0 = blockIdx.x * 32, n0 = blockIdx.y * 32;
  const size_t tb = (size_t)blockIdx.z * CCN;
  const int col = threadIdx.x & 31;
  const int r8  = threadIdx.x >> 5;
#pragma unroll
  for (int s = 0; s < 4; ++s)  // read rows n, contiguous c
    tile[r8 + 8 * s][col] = zin[tb + (size_t)(n0 + r8 + 8 * s) * CC + c0 + col];
  __syncthreads();
#pragma unroll
  for (int s = 0; s < 4; ++s)  // write rows c, contiguous n
    zout[tb + (size_t)(c0 + r8 + 8 * s) * HWN + n0 + col] = tile[col][r8 + 8 * s];
}

// ---------------- depthwise 5x5 conv + BN + residual (z in [c][n]) -------
__global__ __launch_bounds__(256) void k_dwc(
    const _Float16* __restrict__ z_s, const float* __restrict__ dwc_w,
    const float* __restrict__ x, const float* __restrict__ scale4,
    const float* __restrict__ bias4, float* __restrict__ out) {
  __shared__ float sh[20 * 20];
  __shared__ float wsh[25];
  const int plane = blockIdx.x;  // (t*B + b)*C + c
  const int c = plane & (CC - 1);
  const int tid = threadIdx.x;   // h*16 + w
  for (int i = tid; i < 400; i += 256) sh[i] = 0.f;
  if (tid < 25) wsh[tid] = dwc_w[c * 25 + tid];
  __syncthreads();
  const int h = tid >> 4, w = tid & 15;
  sh[(h + 2) * 20 + (w + 2)] = (float)z_s[(size_t)plane * HWN + tid];
  __syncthreads();
  float s = 0.f;
#pragma unroll
  for (int ky = 0; ky < 5; ++ky)
#pragma unroll
    for (int kx = 0; kx < 5; ++kx)
      s += wsh[ky * 5 + kx] * sh[(h + ky) * 20 + (w + kx)];
  size_t o = (size_t)plane * HWN + tid;
  out[o] = scale4[c] * s + bias4[c] + x[o];
}

// ---------------- host-side orchestration ----------------
extern "C" void kernel_launch(void* const* d_in, const int* in_sizes, int n_in,
                              void* d_out, int out_size, void* d_ws,
                              size_t ws_size, hipStream_t stream) {
  const float* x      = (const float*)d_in[0];
  const float* q_w    = (const float*)d_in[1];
  const float* k_w    = (const float*)d_in[2];
  const float* v_w    = (const float*)d_in[3];
  const float* proj_w = (const float*)d_in[4];
  const float* proj_b = (const float*)d_in[5];
  const float* dwc_w  = (const float*)d_in[6];
  const float* dwc_b  = (const float*)d_in[7];
  const float* bn_g   = (const float*)d_in[8];
  const float* bn_b   = (const float*)d_in[9];
  const float* bn_m   = (const float*)d_in[10];
  const float* bn_v   = (const float*)d_in[11];

  char* ws = (char*)d_ws;
  const size_t SPIKE_BYTES = (size_t)TT * TBM * sizeof(_Float16);  // 33.5 MB
  _Float16* xs  = (_Float16*)(ws);                      // reused as o_s
  _Float16* q_s = (_Float16*)(ws + SPIKE_BYTES);        // reused as z_s
  _Float16* k_s = (_Float16*)(ws + 2 * SPIKE_BYTES);    // reused as z_cn
  _Float16* v_s = (_Float16*)(ws + 3 * SPIKE_BYTES);
  _Float16* w16 = (_Float16*)(ws + 4 * SPIKE_BYTES);    // 4 x 512x512 f16
  float* bnsc = (float*)(ws + 4 * SPIKE_BYTES + 4 * CC * CC * sizeof(_Float16));
  float* bnbi = bnsc + 5 * CC;
  _Float16* o_s  = xs;   // xs dead after q/k/v GEMMs
  _Float16* z_s  = q_s;  // q_s dead after attention
  _Float16* z_cn = k_s;  // k_s dead after attention
  float* out = (float*)d_out;

  k_prep_bn<<<(5 * CC + 255) / 256, 256, 0, stream>>>(bn_g, bn_b, bn_m, bn_v,
                                                      proj_b, dwc_b, bnsc, bnbi);
  k_w2h<<<(4 * CC * CC) / 256, 256, 0, stream>>>(q_w, k_w, v_w, proj_w, w16);
  k_lif_in<<<dim3(HWN / 32, CC / 32, BB), 256, 0, stream>>>(x, xs);

  dim3 gg(HWN / 64, CC / 64, BB);
  k_gemm_bn_lif<<<gg, 128, 0, stream>>>(w16 + 0 * CC * CC, xs, q_s, bnsc + 0 * CC,
                                        bnbi + 0 * CC, 1.0f);
  k_gemm_bn_lif<<<gg, 128, 0, stream>>>(w16 + 1 * CC * CC, xs, k_s, bnsc + 1 * CC,
                                        bnbi + 1 * CC, 1.0f);
  k_gemm_bn_lif<<<gg, 128, 0, stream>>>(w16 + 2 * CC * CC, xs, v_s, bnsc + 2 * CC,
                                        bnbi + 2 * CC, 1.0f);

  k_attn<<<dim3(BB, NHD), 256, 0, stream>>>(q_s, k_s, v_s, o_s);

  k_gemm_bn_lif<<<gg, 128, 0, stream>>>(w16 + 3 * CC * CC, o_s, z_s,
                                        bnsc + 3 * CC, bnbi + 3 * CC, 1.0f);

  k_tr_nc_cn<<<dim3(CC / 32, HWN / 32, TT * BB), 256, 0, stream>>>(z_s, z_cn);
  k_dwc<<<TT * BB * CC, 256, 0, stream>>>(z_cn, dwc_w, x, bnsc + 4 * CC,
                                          bnbi + 4 * CC, out);
  (void)in_sizes; (void)n_in; (void)out_size; (void)ws_size;
}
